// RelativeStructureAttention_60198261621495
// MI455X (gfx1250) — matrix operand order repaired and checked by reading, not compiled, not run on hardware
//
#include <hip/hip_runtime.h>

#define DEVINL static __device__ __forceinline__

typedef __attribute__((ext_vector_type(16))) __bf16 bf16x16;
typedef __attribute__((ext_vector_type(8)))  float  f32x8;

union FragAB {
    bf16x16 v;
    uint4 q[2];
    unsigned short h[16];
};

DEVINL unsigned short f32_to_bf16_rte(float f) {
    union { float f; unsigned u; } x; x.f = f;
    unsigned u = x.u;
    unsigned r = u + 0x7fffu + ((u >> 16) & 1u);
    return (unsigned short)(r >> 16);
}

DEVINL float bf16_to_f32(unsigned short u) {
    union { unsigned u; float f; } x; x.u = ((unsigned)u) << 16;
    return x.f;
}

DEVINL f32x8 zero8() {
    f32x8 z;
#pragma unroll
    for (int i = 0; i < 8; ++i) z[i] = 0.f;
    return z;
}

DEVINL f32x8 wmma_bf16(const FragAB& a, const FragAB& b, f32x8 c) {
    return __builtin_amdgcn_wmma_f32_16x16x32_bf16(
        /*neg_a=*/false, a.v, /*neg_b=*/false, b.v,
        /*c_mod=*/(short)0, c, /*reuse_a=*/false, /*reuse_b=*/false);
}

// ---------------------------------------------------------------------------
// Problem constants
// ---------------------------------------------------------------------------
#define BATCH 2
#define SEQ   1024
#define DIM   1024
#define HEADS 16
#define DH    64
#define MROWS (BATCH * SEQ)          // 2048
#define QE_STRIDE 152                // 151 bias entries + 1 pad
#define QE_OFF_BAR   0
#define QE_OFF_POS   17
#define QE_OFF_OCT   113
#define QE_OFF_SEMI  138

// ---------------------------------------------------------------------------
// K1a: f32 -> bf16 conversion (grid-stride)
// ---------------------------------------------------------------------------
__global__ void k_cvt_bf16(const float* __restrict__ src,
                           unsigned short* __restrict__ dst, int n) {
    for (int i = blockIdx.x * blockDim.x + threadIdx.x; i < n;
         i += gridDim.x * blockDim.x)
        dst[i] = f32_to_bf16_rte(src[i]);
}

// ---------------------------------------------------------------------------
// K1b: build combined structure-bias table, f32 [151][64]
// rows: bar(16)+z, pos(95)+z, oct(24)+z, semi(12)+z
// ---------------------------------------------------------------------------
__global__ void k_build_table(const float* __restrict__ bar,
                              const float* __restrict__ pos,
                              const float* __restrict__ oct,
                              const float* __restrict__ semi,
                              float* __restrict__ table) {
    int t = blockIdx.x * blockDim.x + threadIdx.x;
    if (t >= 151 * 64) return;
    int s = t / 64, d = t % 64;
    float v = 0.f;
    if (s < 16)        v = bar[s * 64 + d];
    else if (s == 16)  v = 0.f;
    else if (s < 112)  v = pos[(s - 17) * 64 + d];
    else if (s == 112) v = 0.f;
    else if (s < 137)  v = oct[(s - 113) * 64 + d];
    else if (s == 137) v = 0.f;
    else if (s < 150)  v = semi[(s - 138) * 64 + d];
    table[t] = v;
}

// ---------------------------------------------------------------------------
// K2: projection GEMM  C(2048x1024) = A(2048x1024) * W^T (W is N x K, bf16)
// One wave computes a 16x64 output tile with 4 f32 WMMA accumulators.
// grid = (16 n-blocks == heads, 16 m-groups), block = 256 (8 waves).
// mode 0/1: store bf16 as [b][h][i][d]   (q / k)
// mode 2  : store bf16 transposed [b][h][d][i]  (v^T, for PV B-fragments)
// ---------------------------------------------------------------------------
__global__ void k_gemm_qkv(const unsigned short* __restrict__ A,
                           const unsigned short* __restrict__ W,
                           unsigned short* __restrict__ dst, int mode) {
    const int lane = threadIdx.x & 31;
    const int wv   = threadIdx.x >> 5;
    const int Hh   = lane >> 4;
    const int nloc = lane & 15;
    const int head = blockIdx.x;                 // n-block of 64 == head
    const int m0   = (blockIdx.y * 8 + wv) * 16; // 128 m-tiles total
    const int mrow = m0 + (lane & 15);

    f32x8 acc[4];
#pragma unroll
    for (int c = 0; c < 4; ++c) acc[c] = zero8();

    const unsigned short* arow = A + (size_t)mrow * DIM;
    for (int k0 = 0; k0 < DIM; k0 += 32) {
        FragAB a;
        a.q[0] = *(const uint4*)(arow + k0 + 16 * Hh);
        a.q[1] = *(const uint4*)(arow + k0 + 16 * Hh + 8);
#pragma unroll
        for (int c = 0; c < 4; ++c) {
            const unsigned short* wr =
                W + (size_t)(head * 64 + c * 16 + nloc) * DIM + k0 + 16 * Hh;
            FragAB b;
            b.q[0] = *(const uint4*)(wr);
            b.q[1] = *(const uint4*)(wr + 8);
            acc[c] = wmma_bf16(a, b, acc[c]);
        }
    }

#pragma unroll
    for (int c = 0; c < 4; ++c) {
#pragma unroll
        for (int g = 0; g < 8; ++g) {
            int mloc = g + 8 * Hh;
            int m = m0 + mloc;
            int bb = m >> 10, ii = m & (SEQ - 1);
            int dc = c * 16 + nloc;
            unsigned short val = f32_to_bf16_rte(acc[c][g]);
            if (mode == 2)
                dst[(((size_t)bb * HEADS + head) * DH + dc) * SEQ + ii] = val;
            else
                dst[(((size_t)bb * HEADS + head) * SEQ + ii) * DH + dc] = val;
        }
    }
}

// ---------------------------------------------------------------------------
// K3: QE[b,h,i,s] = sum_d q[b,h,i,d] * table[s,d]   (s = 0..150)
// one block per (b,h,i) row
// ---------------------------------------------------------------------------
__global__ void k_qe(const unsigned short* __restrict__ qb,
                     const float* __restrict__ table,
                     float* __restrict__ qe) {
    __shared__ float qrow[DH];
    const int row = blockIdx.x;          // 0 .. B*H*SEQ-1
    const int t = threadIdx.x;
    if (t < DH) qrow[t] = bf16_to_f32(qb[(size_t)row * DH + t]);
    __syncthreads();
    if (t < 151) {
        float s = 0.f;
#pragma unroll
        for (int d = 0; d < DH; ++d) s += qrow[d] * table[t * DH + d];
        qe[(size_t)row * QE_STRIDE + t] = s;
    }
}

// ---------------------------------------------------------------------------
// K4: fused flash attention with relative-structure biases.
// grid = (64 i-tiles, 2 batches), block = 512 (16 waves == 16 heads).
// LDS: QE rows for all heads (156KB) + shared idx tiles (8KB) + P bounce (16KB)
// ---------------------------------------------------------------------------
#define LDS_QE_BYTES  (HEADS * 16 * QE_STRIDE * 4)   // 155648
#define LDS_IDX_BYTES (4 * 16 * 32 * 4)              // 8192
#define LDS_P_BYTES   (HEADS * 16 * 32 * 2)          // 16384
#define LDS_TOTAL     (LDS_QE_BYTES + LDS_IDX_BYTES + LDS_P_BYTES)

__global__ __launch_bounds__(512, 1)
void k_attn(const unsigned short* __restrict__ qbuf,   // [b][h][n][d]
            const unsigned short* __restrict__ kbuf,   // [b][h][n][d]
            const unsigned short* __restrict__ vtbuf,  // [b][h][d][n]
            const unsigned short* __restrict__ erbuf,  // [n][d]
            const float* __restrict__ qe,              // [b][h][n][152]
            const int* __restrict__ idx_bar,
            const int* __restrict__ idx_pos,
            const int* __restrict__ idx_oct,
            const int* __restrict__ idx_semi,
            unsigned short* __restrict__ attn_out) {   // [b][n][dim] bf16
    extern __shared__ char smem[];
    float* lds_qe = (float*)smem;
    int* lds_idx = (int*)(smem + LDS_QE_BYTES);
    unsigned short* lds_p = (unsigned short*)(smem + LDS_QE_BYTES + LDS_IDX_BYTES);

    const int tid = threadIdx.x;
    const int lane = tid & 31;
    const int h = tid >> 5;          // wave == head
    const int Hh = lane >> 4;
    const int nloc = lane & 15;
    const int i0 = blockIdx.x * 16;
    const int b = blockIdx.y;

    // stage QE rows for this (b, i-tile), all heads
    {
        const float* src = qe + ((size_t)b * HEADS * SEQ) * QE_STRIDE;
        for (int t = tid; t < HEADS * 16 * QE_STRIDE; t += 512) {
            int s = t % QE_STRIDE;
            int r = (t / QE_STRIDE) & 15;
            int hh = t / (QE_STRIDE * 16);
            lds_qe[t] = src[((size_t)hh * SEQ + i0 + r) * QE_STRIDE + s];
        }
    }

    // Q fragments (A-layout, 16x64 = two 16x32 frags)
    const unsigned short* qbase = qbuf + (((size_t)b * HEADS + h) * SEQ + i0) * DH;
    FragAB a0, a1;
    {
        const unsigned short* qr = qbase + (size_t)(lane & 15) * DH;
        a0.q[0] = *(const uint4*)(qr + 16 * Hh);
        a0.q[1] = *(const uint4*)(qr + 16 * Hh + 8);
        a1.q[0] = *(const uint4*)(qr + 32 + 16 * Hh);
        a1.q[1] = *(const uint4*)(qr + 32 + 16 * Hh + 8);
    }
    const unsigned short* kbase = kbuf + (((size_t)b * HEADS + h) * SEQ) * DH;
    const unsigned short* vbase = vtbuf + (((size_t)b * HEADS + h) * DH) * SEQ;

    f32x8 o[4];
#pragma unroll
    for (int c = 0; c < 4; ++c) o[c] = zero8();
    float m_r[8], l_r[8];
#pragma unroll
    for (int g = 0; g < 8; ++g) { m_r[g] = -3.0e38f; l_r[g] = 0.f; }

    const int jend = i0 + 15;
    for (int j0 = 0; j0 <= jend; j0 += 32) {
        __syncthreads();   // previous iteration done with lds_idx
        {
            int r = tid >> 5, cj = tid & 31;
            size_t gi = ((size_t)b * SEQ + i0 + r) * SEQ + j0 + cj;
            lds_idx[(0 * 16 + r) * 32 + cj] = idx_bar[gi];
            lds_idx[(1 * 16 + r) * 32 + cj] = idx_pos[gi];
            lds_idx[(2 * 16 + r) * 32 + cj] = idx_oct[gi];
            lds_idx[(3 * 16 + r) * 32 + cj] = idx_semi[gi];
        }
        __syncthreads();

        // prefetch next iteration's K/V rows into cache
        if (j0 + 32 <= jend) {
            __builtin_prefetch(kbase + (size_t)(j0 + 32 + lane) * DH, 0, 1);
            __builtin_prefetch(vbase + (size_t)nloc * SEQ + j0 + 32, 0, 1);
        }

        // ---- S = Q K^T (two 16x16 subtiles over 32 keys) ----
        f32x8 s[2];
        s[0] = zero8(); s[1] = zero8();
#pragma unroll
        for (int sub = 0; sub < 2; ++sub) {
            const unsigned short* kr = kbase + (size_t)(j0 + sub * 16 + nloc) * DH;
            FragAB b0, b1;
            b0.q[0] = *(const uint4*)(kr + 16 * Hh);
            b0.q[1] = *(const uint4*)(kr + 16 * Hh + 8);
            b1.q[0] = *(const uint4*)(kr + 32 + 16 * Hh);
            b1.q[1] = *(const uint4*)(kr + 32 + 16 * Hh + 8);
            s[sub] = wmma_bf16(a0, b0, s[sub]);
            s[sub] = wmma_bf16(a1, b1, s[sub]);
        }

        // ---- P_er = Q @ Er[base .. base+47]^T  (skew band) ----
        f32x8 per[3];
        const int base = SEQ - 16 - i0 + j0;   // >= 0 always
#pragma unroll
        for (int sub = 0; sub < 3; ++sub) {
            int rrow = base + sub * 16 + nloc;
            rrow = (rrow < SEQ - 1) ? rrow : (SEQ - 1);  // clamp (masked region)
            const unsigned short* er = erbuf + (size_t)rrow * DH;
            FragAB e0, e1;
            e0.q[0] = *(const uint4*)(er + 16 * Hh);
            e0.q[1] = *(const uint4*)(er + 16 * Hh + 8);
            e1.q[0] = *(const uint4*)(er + 32 + 16 * Hh);
            e1.q[1] = *(const uint4*)(er + 32 + 16 * Hh + 8);
            per[sub] = zero8();
            per[sub] = wmma_bf16(a0, e0, per[sub]);
            per[sub] = wmma_bf16(a1, e1, per[sub]);
        }

        // ---- combine Srel (diagonal extract via bpermute) + biases, mask ----
        float tmax[8];
#pragma unroll
        for (int g = 0; g < 8; ++g) tmax[g] = -3.0e38f;
#pragma unroll
        for (int g = 0; g < 8; ++g) {
            const int mi = g + 8 * Hh;
            const int tsub = 15 - mi + nloc;                 // 0..30
            const int srcLane = (lane & 16) | (tsub & 15);
            const float* qrow = lds_qe + ((size_t)h * 16 + mi) * QE_STRIDE;
#pragma unroll
            for (int sub = 0; sub < 2; ++sub) {
                float lo = __shfl(per[sub][g], srcLane, 32);
                float hi = __shfl(per[sub + 1][g], srcLane, 32);
                float srel = (tsub < 16) ? lo : hi;
                int jb = sub * 16 + nloc;
                int t0 = lds_idx[(0 * 16 + mi) * 32 + jb];
                int t1 = lds_idx[(1 * 16 + mi) * 32 + jb];
                int t2 = lds_idx[(2 * 16 + mi) * 32 + jb];
                int t3 = lds_idx[(3 * 16 + mi) * 32 + jb];
                float bias = qrow[QE_OFF_BAR + t0] + qrow[QE_OFF_POS + t1] +
                             qrow[QE_OFF_OCT + t2] + qrow[QE_OFF_SEMI + t3];
                float sv = (s[sub][g] + srel + bias) * 0.125f;
                bool ok = (j0 + jb) <= (i0 + mi);
                sv = ok ? sv : -1.0e9f;
                s[sub][g] = sv;
                tmax[g] = fmaxf(tmax[g], sv);
            }
        }

        // ---- online softmax (row reductions inside 16-lane half) ----
#pragma unroll
        for (int g = 0; g < 8; ++g) {
            float v = tmax[g];
            v = fmaxf(v, __shfl_xor(v, 1, 32));
            v = fmaxf(v, __shfl_xor(v, 2, 32));
            v = fmaxf(v, __shfl_xor(v, 4, 32));
            v = fmaxf(v, __shfl_xor(v, 8, 32));
            tmax[g] = v;
        }
        float corr[8];
#pragma unroll
        for (int g = 0; g < 8; ++g) {
            float mnew = fmaxf(m_r[g], tmax[g]);
            corr[g] = __expf(m_r[g] - mnew);
            m_r[g] = mnew;
        }
        float tsum[8];
#pragma unroll
        for (int g = 0; g < 8; ++g) tsum[g] = 0.f;
#pragma unroll
        for (int sub = 0; sub < 2; ++sub) {
#pragma unroll
            for (int g = 0; g < 8; ++g) {
                int mi = g + 8 * Hh;
                float p = __expf(s[sub][g] - m_r[g]);
                tsum[g] += p;
                lds_p[((size_t)h * 16 + mi) * 32 + sub * 16 + nloc] =
                    f32_to_bf16_rte(p);
            }
        }
#pragma unroll
        for (int g = 0; g < 8; ++g) {
            float v = tsum[g];
            v += __shfl_xor(v, 1, 32);
            v += __shfl_xor(v, 2, 32);
            v += __shfl_xor(v, 4, 32);
            v += __shfl_xor(v, 8, 32);
            l_r[g] = l_r[g] * corr[g] + v;
        }
#pragma unroll
        for (int c = 0; c < 4; ++c)
#pragma unroll
            for (int g = 0; g < 8; ++g) o[c][g] *= corr[g];

        // ---- P (C-layout) -> A-layout via per-wave LDS bounce ----
        FragAB pa;
        {
            const unsigned short* pr = lds_p + ((size_t)h * 16 + (lane & 15)) * 32;
            pa.q[0] = *(const uint4*)(pr + 16 * Hh);
            pa.q[1] = *(const uint4*)(pr + 16 * Hh + 8);
        }

        // ---- O += P @ V  (V pre-transposed: rows contiguous in j) ----
#pragma unroll
        for (int c = 0; c < 4; ++c) {
            const unsigned short* vr =
                vbase + (size_t)(c * 16 + nloc) * SEQ + j0 + 16 * Hh;
            FragAB vb;
            vb.q[0] = *(const uint4*)(vr);
            vb.q[1] = *(const uint4*)(vr + 8);
            o[c] = wmma_bf16(pa, vb, o[c]);
        }
    }

    // epilogue: divide by row sums, store bf16 [b][i][h*64+dc]
#pragma unroll
    for (int c = 0; c < 4; ++c) {
#pragma unroll
        for (int g = 0; g < 8; ++g) {
            int mi = g + 8 * Hh;
            float v = o[c][g] / l_r[g];
            attn_out[((size_t)b * SEQ + i0 + mi) * DIM + h * DH + c * 16 + nloc] =
                f32_to_bf16_rte(v);
        }
    }
}

// ---------------------------------------------------------------------------
// K5: output projection  out(2048x1024 f32) = attn(bf16) @ Wo^T(bf16)
// ---------------------------------------------------------------------------
__global__ void k_gemm_out(const unsigned short* __restrict__ A,
                           const unsigned short* __restrict__ W,
                           float* __restrict__ dst) {
    const int lane = threadIdx.x & 31;
    const int wv   = threadIdx.x >> 5;
    const int Hh   = lane >> 4;
    const int nloc = lane & 15;
    const int nblk = blockIdx.x;                 // 16 blocks of 64 cols
    const int m0   = (blockIdx.y * 8 + wv) * 16;

    f32x8 acc[4];
#pragma unroll
    for (int c = 0; c < 4; ++c) acc[c] = zero8();

    const unsigned short* arow = A + (size_t)(m0 + (lane & 15)) * DIM;
    for (int k0 = 0; k0 < DIM; k0 += 32) {
        FragAB a;
        a.q[0] = *(const uint4*)(arow + k0 + 16 * Hh);
        a.q[1] = *(const uint4*)(arow + k0 + 16 * Hh + 8);
#pragma unroll
        for (int c = 0; c < 4; ++c) {
            const unsigned short* wr =
                W + (size_t)(nblk * 64 + c * 16 + nloc) * DIM + k0 + 16 * Hh;
            FragAB b;
            b.q[0] = *(const uint4*)(wr);
            b.q[1] = *(const uint4*)(wr + 8);
            acc[c] = wmma_bf16(a, b, acc[c]);
        }
    }
#pragma unroll
    for (int c = 0; c < 4; ++c)
#pragma unroll
        for (int g = 0; g < 8; ++g) {
            int m = m0 + g + 8 * Hh;
            dst[(size_t)m * DIM + nblk * 64 + c * 16 + nloc] = acc[c][g];
        }
}

// ---------------------------------------------------------------------------
// host launcher
// ---------------------------------------------------------------------------
extern "C" void kernel_launch(void* const* d_in, const int* in_sizes, int n_in,
                              void* d_out, int out_size, void* d_ws, size_t ws_size,
                              hipStream_t stream) {
    (void)in_sizes; (void)n_in; (void)out_size; (void)ws_size;
    const float* x    = (const float*)d_in[0];
    const int*   rbar = (const int*)d_in[1];
    const int*   rpos = (const int*)d_in[2];
    const int*   roct = (const int*)d_in[3];
    const int*   rsem = (const int*)d_in[4];
    const float* Wq   = (const float*)d_in[5];
    const float* Wk   = (const float*)d_in[6];
    const float* Wv   = (const float*)d_in[7];
    const float* Wo   = (const float*)d_in[8];
    const float* Er   = (const float*)d_in[9];
    const float* Ebar = (const float*)d_in[10];
    const float* Epos = (const float*)d_in[11];
    const float* Eoct = (const float*)d_in[12];
    const float* Esem = (const float*)d_in[13];
    float* out = (float*)d_out;

    char* ws = (char*)d_ws;
    size_t off = 0;
    auto take = [&](size_t bytes) -> char* {
        char* p = ws + off;
        off = (off + bytes + 255) & ~(size_t)255;
        return p;
    };
    unsigned short* xb   = (unsigned short*)take((size_t)MROWS * DIM * 2);
    unsigned short* wqb  = (unsigned short*)take((size_t)DIM * DIM * 2);
    unsigned short* wkb  = (unsigned short*)take((size_t)DIM * DIM * 2);
    unsigned short* wvb  = (unsigned short*)take((size_t)DIM * DIM * 2);
    unsigned short* wob  = (unsigned short*)take((size_t)DIM * DIM * 2);
    unsigned short* erb  = (unsigned short*)take((size_t)SEQ * DH * 2);
    float*          tab  = (float*)take((size_t)151 * DH * 4);
    unsigned short* qbuf = (unsigned short*)take((size_t)MROWS * DIM * 2);
    unsigned short* kbuf = (unsigned short*)take((size_t)MROWS * DIM * 2);
    unsigned short* vtb  = (unsigned short*)take((size_t)MROWS * DIM * 2);
    float*          qe   = (float*)take((size_t)BATCH * HEADS * SEQ * QE_STRIDE * 4);
    unsigned short* attn = (unsigned short*)take((size_t)MROWS * DIM * 2);

    // conversions
    k_cvt_bf16<<<2048, 256, 0, stream>>>(x, xb, MROWS * DIM);
    k_cvt_bf16<<<1024, 256, 0, stream>>>(Wq, wqb, DIM * DIM);
    k_cvt_bf16<<<1024, 256, 0, stream>>>(Wk, wkb, DIM * DIM);
    k_cvt_bf16<<<1024, 256, 0, stream>>>(Wv, wvb, DIM * DIM);
    k_cvt_bf16<<<1024, 256, 0, stream>>>(Wo, wob, DIM * DIM);
    k_cvt_bf16<<<256, 256, 0, stream>>>(Er, erb, SEQ * DH);
    k_build_table<<<(151 * 64 + 255) / 256, 256, 0, stream>>>(Ebar, Epos, Eoct, Esem, tab);

    // projections (V stored transposed)
    dim3 gg(16, 16);
    k_gemm_qkv<<<gg, 256, 0, stream>>>(xb, wqb, qbuf, 0);
    k_gemm_qkv<<<gg, 256, 0, stream>>>(xb, wkb, kbuf, 1);
    k_gemm_qkv<<<gg, 256, 0, stream>>>(xb, wvb, vtb, 2);

    // per-row structure-bias tables
    k_qe<<<BATCH * HEADS * SEQ, 160, 0, stream>>>(qbuf, tab, qe);

    // fused flash attention: 16 waves == 16 heads per workgroup
    dim3 ga(SEQ / 16, BATCH);
    k_attn<<<ga, 512, LDS_TOTAL, stream>>>(qbuf, kbuf, vtb, erb, qe,
                                           rbar, rpos, roct, rsem, attn);

    // output projection
    k_gemm_out<<<gg, 256, 0, stream>>>(attn, wob, out);
}
